// KernelSVM_16707422782018
// MI455X (gfx1250) — compile-verified, matrix operand order checked by reading, not run on hardware
//
#include <hip/hip_runtime.h>
#include <hip/hip_bf16.h>

typedef __attribute__((ext_vector_type(16))) __bf16 v16bf;
typedef __attribute__((ext_vector_type(8)))  float  v8f;

#define RBF_N 524288
#define RBF_D 64
#define RBF_H 256
#define ROWS_PER_BLOCK 128   // 8 waves * 16 rows
#define C_STRIDE 72          // 64 bf16 + 8 pad (keeps 16B alignment, spreads banks)

__global__ __launch_bounds__(256)
void rbf_head_wmma(const float* __restrict__ X,
                   const float* __restrict__ Cent,
                   const float* __restrict__ Wv,
                   const float* __restrict__ Bv,
                   float* __restrict__ Out)
{
    __shared__ __bf16 sC[RBF_H * C_STRIDE];  // centers, bf16, row-major padded
    __shared__ float  sc2[RBF_H];            // ||c_h||^2
    __shared__ float  sw[RBF_H];             // w[h]

    const int tid  = threadIdx.x;
    const int lane = tid & 31;
    const int wave = tid >> 5;
    const int lr   = lane & 15;   // row-in-tile / col-in-tile index
    const int hf   = lane >> 4;   // which K-half this lane carries

    // ---- cooperative: centers fp32 -> bf16 LDS, c2, w (one row per thread) ----
    {
        const float* cr = Cent + (size_t)tid * RBF_D;
        float s = 0.f;
        #pragma unroll
        for (int k = 0; k < RBF_D; ++k) {
            float v = cr[k];
            s = fmaf(v, v, s);
            sC[tid * C_STRIDE + k] = (__bf16)v;
        }
        sc2[tid] = s;
        sw[tid]  = Wv[tid];
    }
    __syncthreads();

    // ---- per-wave: 16 rows of X -> two resident bf16 A fragments ----
    const int row0 = blockIdx.x * ROWS_PER_BLOCK + wave * 16;
    v16bf a0, a1;            // K = 0..31 and K = 32..63 (lane-split K-halves)
    float xp = 0.f;          // partial sum of squares for this lane's 32 values
    {
        const float* xr = X + (size_t)(row0 + lr) * RBF_D + hf * 16;
        #pragma unroll
        for (int j = 0; j < 16; ++j) {
            float v0 = xr[j];
            float v1 = xr[32 + j];
            a0[j] = (__bf16)v0;
            a1[j] = (__bf16)v1;
            xp = fmaf(v0, v0, fmaf(v1, v1, xp));
        }
    }
    // combine the two K-halves of each row: lane L and L^16 hold the same row
    xp += __shfl_xor(xp, 16);
    // x2 for the 8 accumulator slots: slot i maps to row (i + 8*hf)
    float x2s[8];
    #pragma unroll
    for (int i = 0; i < 8; ++i)
        x2s[i] = __shfl(xp, i + 8 * hf);

    float rowacc[8];
    #pragma unroll
    for (int i = 0; i < 8; ++i) rowacc[i] = 0.f;

    // ---- loop over 16 column tiles of H; fuse exp + weighted sum per tile ----
    for (int t = 0; t < 16; ++t) {
        const __bf16* bp = &sC[(t * 16 + lr) * C_STRIDE + hf * 16];
        v16bf b0, b1;
        #pragma unroll
        for (int j = 0; j < 16; ++j) {
            b0[j] = bp[j];        // K = 0..31 half
            b1[j] = bp[32 + j];   // K = 32..63 half
        }

        v8f acc = {};
        acc = __builtin_amdgcn_wmma_f32_16x16x32_bf16(
                  false, a0, false, b0, (short)0, acc, false, false);
        acc = __builtin_amdgcn_wmma_f32_16x16x32_bf16(
                  false, a1, false, b1, (short)0, acc, false, false);

        const float c2n = sc2[t * 16 + lr];   // this lane's column n = t*16 + lr
        const float wn  = sw[t * 16 + lr];
        #pragma unroll
        for (int i = 0; i < 8; ++i) {
            // d2 = max(x2[m] - 2*xc + c2[n], 0);  feat = exp(-d2)
            float d2 = fmaf(-2.f, acc[i], x2s[i] + c2n);
            d2 = fmaxf(d2, 0.f);
            rowacc[i] = fmaf(wn, __expf(-d2), rowacc[i]);
        }
    }

    // ---- reduce across the 16 column-lanes of each half; write 16 rows ----
    const float bias = Bv[0];
    #pragma unroll
    for (int i = 0; i < 8; ++i) {
        float v = rowacc[i];
        v += __shfl_xor(v, 1);
        v += __shfl_xor(v, 2);
        v += __shfl_xor(v, 4);
        v += __shfl_xor(v, 8);
        if (lr == 0)
            Out[(size_t)row0 + i + 8 * hf] = v + bias;
    }
}

extern "C" void kernel_launch(void* const* d_in, const int* in_sizes, int n_in,
                              void* d_out, int out_size, void* d_ws, size_t ws_size,
                              hipStream_t stream) {
    (void)in_sizes; (void)n_in; (void)d_ws; (void)ws_size; (void)out_size;
    const float* X  = (const float*)d_in[0];   // [N, 64]
    const float* C  = (const float*)d_in[1];   // [256, 64]
    const float* W  = (const float*)d_in[2];   // [1, 256]
    const float* B  = (const float*)d_in[3];   // [1]
    float* Out      = (float*)d_out;           // [N, 1]

    dim3 grid(RBF_N / ROWS_PER_BLOCK);         // 4096 blocks
    dim3 block(256);                           // 8 waves (wave32)
    rbf_head_wmma<<<grid, block, 0, stream>>>(X, C, W, B, Out);
}
